// NodeGAT_39350490365954
// MI455X (gfx1250) — matrix operand order repaired and checked by reading, not compiled, not run on hardware
//
#include <hip/hip_runtime.h>

// ---------------------------------------------------------------------------
// GATv2 (3 layers) for MI455X / gfx1250.
//  * Fused dual GEMM (xl = x@Wl^T+bl, xr = x@Wr^T+br) on the matrix pipe:
//    v_wmma_f32_16x16x32_bf16. Each wave32 owns a 16-row strip x 4 col tiles
//    for BOTH weight matrices -> one A-fragment load feeds 8 WMMAs.
//    Layer-3 weights are zero-padded to 48 cols so the K-loop has no
//    divergent guards (EXEC stays all-ones around every WMMA).
//  * Edge phase is HBM-bound: float4 gathers, wave-level head reduction,
//    ordered-uint atomicMax for segment max, native f32 atomic adds.
// ---------------------------------------------------------------------------

typedef __attribute__((ext_vector_type(16))) __bf16 v16bf;
typedef __attribute__((ext_vector_type(8)))  __bf16 v8bf;
typedef __attribute__((ext_vector_type(8)))  float  v8f;

#define NEG_SLOPE 0.2f

__device__ __forceinline__ unsigned short f32_to_bf16(float f) {
  unsigned u = __float_as_uint(f);
  unsigned rounding = 0x7FFFu + ((u >> 16) & 1u);   // round-to-nearest-even
  return (unsigned short)((u + rounding) >> 16);
}

// order-preserving float <-> uint mapping (for atomicMax on floats)
__device__ __forceinline__ unsigned f2ord(float f) {
  unsigned u = __float_as_uint(f);
  return (u & 0x80000000u) ? ~u : (u | 0x80000000u);
}
__device__ __forceinline__ float ord2f(unsigned u) {
  unsigned v = (u & 0x80000000u) ? (u & 0x7FFFFFFFu) : ~u;
  return __uint_as_float(v);
}
#define ORD_NEG_INF 0x007FFFFFu   // f2ord(-inf)

// ----------------------------- utility kernels -----------------------------

__global__ void cast_f32_bf16_k(const float* __restrict__ in,
                                unsigned short* __restrict__ out, int n) {
  int t = blockIdx.x * blockDim.x + threadIdx.x;
  if (t < n) out[t] = f32_to_bf16(in[t]);
}

__global__ void fill_f32_k(float* __restrict__ p, float v, int n) {
  int t = blockIdx.x * blockDim.x + threadIdx.x;
  if (t < n) p[t] = v;
}

__global__ void fill_u32_k(unsigned* __restrict__ p, unsigned v, int n) {
  int t = blockIdx.x * blockDim.x + threadIdx.x;
  if (t < n) p[t] = v;
}

// ------------------------------ WMMA GEMM ----------------------------------
// Outl[n,j] = sum_k A[n,k]*Wl[j,k] + bl[j];  Outr likewise.
// A: N x 128 bf16 row-major. Wl/Wr: (padded Mout) x 128 bf16 row-major, so
// B[k][j] = W[j*128+k] gives contiguous 16-element K runs per lane (column
// per lane). One wave: 16 rows x COLT col tiles x 2 outputs; K=128 in 4
// steps -> 8*COLT WMMAs per wave, one A fragment load per K step.

template <int COLT>
__global__ void gemm_dual_bf16_wmma_k(const unsigned short* __restrict__ Abf,
                                      const unsigned short* __restrict__ Wlbf,
                                      const unsigned short* __restrict__ Wrbf,
                                      const float* __restrict__ bl,
                                      const float* __restrict__ br,
                                      float* __restrict__ Outl,
                                      float* __restrict__ Outr,
                                      int N, int Mout, int rowTiles) {
  const int K = 128;
  int lane = threadIdx.x & 31;
  int tm   = blockIdx.x * (blockDim.x >> 5) + (threadIdx.x >> 5);
  if (tm >= rowTiles) return;                 // wave-uniform exit
  int tnBase = blockIdx.y * COLT;

  int colLane = lane & 15;
  int rowG = tm * 16 + colLane;
  int rowL = rowG < N ? rowG : (N - 1);       // clamp loads; stores guarded
  int kbA  = (lane < 16) ? 0 : 16;
  int kbB  = (lane < 16) ? 0 : 16;

  const __bf16* Arow = (const __bf16*)Abf + (size_t)rowL * K;
  const __bf16* Bl   = (const __bf16*)Wlbf + ((size_t)(tnBase * 16) + colLane) * K + kbB;
  const __bf16* Br   = (const __bf16*)Wrbf + ((size_t)(tnBase * 16) + colLane) * K + kbB;

  v8f cl[COLT] = {};
  v8f cr[COLT] = {};
#pragma unroll
  for (int k0 = 0; k0 < K; k0 += 32) {
    v16bf Af;
    {
      v8bf a0 = *(const v8bf*)(Arow + k0 + kbA);       // K = k0+kb .. +7
      v8bf a1 = *(const v8bf*)(Arow + k0 + 8 + kbA);   // K = k0+8+kb .. +7
#pragma unroll
      for (int i = 0; i < 8; ++i) { Af[i] = a0[i]; Af[i + 8] = a1[i]; }
    }
#pragma unroll
    for (int t = 0; t < COLT; ++t) {
      v16bf Bf;
      {
        v8bf b0 = *(const v8bf*)(Bl + (size_t)t * 16 * K + k0);
        v8bf b1 = *(const v8bf*)(Bl + (size_t)t * 16 * K + k0 + 8);
#pragma unroll
        for (int i = 0; i < 8; ++i) { Bf[i] = b0[i]; Bf[i + 8] = b1[i]; }
      }
      cl[t] = __builtin_amdgcn_wmma_f32_16x16x32_bf16(false, Af, false, Bf,
                                                      (short)0, cl[t], false, false);
      {
        v8bf b0 = *(const v8bf*)(Br + (size_t)t * 16 * K + k0);
        v8bf b1 = *(const v8bf*)(Br + (size_t)t * 16 * K + k0 + 8);
#pragma unroll
        for (int i = 0; i < 8; ++i) { Bf[i] = b0[i]; Bf[i + 8] = b1[i]; }
      }
      cr[t] = __builtin_amdgcn_wmma_f32_16x16x32_bf16(false, Af, false, Bf,
                                                      (short)0, cr[t], false, false);
    }
  }

  // D layout: lane -> column (lane&15); VGPR v -> row v (+8 for hi half-wave)
  int mbase = (lane < 16) ? 0 : 8;
#pragma unroll
  for (int t = 0; t < COLT; ++t) {
    int colG = (tnBase + t) * 16 + colLane;
    if (colG < Mout) {
      float bvl = bl[colG];
      float bvr = br[colG];
#pragma unroll
      for (int v = 0; v < 8; ++v) {
        int mg = tm * 16 + mbase + v;
        if (mg < N) {
          Outl[(size_t)mg * Mout + colG] = cl[t][v] + bvl;
          Outr[(size_t)mg * Mout + colG] = cr[t][v] + bvr;
        }
      }
    }
  }
}

// ------------------------- edge kernels, layers 1-2 ------------------------
// H=4, C=32, HC=128.  One wave per edge; lane covers 4 features (float4);
// h = lane>>3; reduce each 8-lane group to the per-head score.

__global__ void edge_score_h4_k(const float* __restrict__ xl,
                                const float* __restrict__ xr,
                                const int* __restrict__ eidx, int E, int E2,
                                const float* __restrict__ att,   // [128]
                                float* __restrict__ score,       // [E2*4]
                                unsigned* __restrict__ mOrd) {   // [N*4]
  int wave = (blockIdx.x * blockDim.x + threadIdx.x) >> 5;
  int lane = threadIdx.x & 31;
  if (wave >= E2) return;
  int src, dst;
  if (wave < E) { src = eidx[wave]; dst = eidx[E + wave]; }
  else          { src = wave - E;   dst = src; }

  float4 a = *(const float4*)(xl + (size_t)src * 128 + lane * 4);
  float4 b = *(const float4*)(xr + (size_t)dst * 128 + lane * 4);
  float4 w = *(const float4*)(att + lane * 4);
  float p = 0.f, e;
  e = a.x + b.x; e = e > 0.f ? e : NEG_SLOPE * e; p += w.x * e;
  e = a.y + b.y; e = e > 0.f ? e : NEG_SLOPE * e; p += w.y * e;
  e = a.z + b.z; e = e > 0.f ? e : NEG_SLOPE * e; p += w.z * e;
  e = a.w + b.w; e = e > 0.f ? e : NEG_SLOPE * e; p += w.w * e;
  p += __shfl_xor(p, 1, 32);
  p += __shfl_xor(p, 2, 32);
  p += __shfl_xor(p, 4, 32);
  if ((lane & 7) == 0) {
    int h = lane >> 3;
    score[(size_t)wave * 4 + h] = p;
    atomicMax(&mOrd[(size_t)dst * 4 + h], f2ord(p));
  }
}

__global__ void edge_exp_k(float* __restrict__ score,
                           const int* __restrict__ eidx, int E, int E2, int H,
                           const unsigned* __restrict__ mOrd,
                           float* __restrict__ ssum) {
  int tid = blockIdx.x * blockDim.x + threadIdx.x;
  if (tid >= E2 * H) return;
  int e = tid / H, h = tid - e * H;
  int dst = (e < E) ? eidx[E + e] : (e - E);
  float m = ord2f(mOrd[(size_t)dst * H + h]);
  float p = __expf(score[tid] - m);
  score[tid] = p;
  unsafeAtomicAdd(&ssum[(size_t)dst * H + h], p);
}

__global__ void edge_agg_h4_k(const float* __restrict__ xl,
                              const int* __restrict__ eidx, int E, int E2,
                              const float* __restrict__ score,  // p values
                              const float* __restrict__ ssum,
                              float* __restrict__ acc) {
  int wave = (blockIdx.x * blockDim.x + threadIdx.x) >> 5;
  int lane = threadIdx.x & 31;
  if (wave >= E2) return;
  int src, dst;
  if (wave < E) { src = eidx[wave]; dst = eidx[E + wave]; }
  else          { src = wave - E;   dst = src; }
  int h = lane >> 3;
  float alpha = score[(size_t)wave * 4 + h] / ssum[(size_t)dst * 4 + h];
  float4 v = *(const float4*)(xl + (size_t)src * 128 + lane * 4);
  float* o = acc + (size_t)dst * 128 + lane * 4;
  unsafeAtomicAdd(o + 0, alpha * v.x);
  unsafeAtomicAdd(o + 1, alpha * v.y);
  unsafeAtomicAdd(o + 2, alpha * v.z);
  unsafeAtomicAdd(o + 3, alpha * v.w);
}

__global__ void bias_elu_k(float* __restrict__ acc,
                           const float* __restrict__ bias, int n) {
  int t = blockIdx.x * blockDim.x + threadIdx.x;
  if (t < n) {
    float v = acc[t] + bias[t & 127];
    acc[t] = v > 0.f ? v : (__expf(v) - 1.f);
  }
}

// --------------------------- edge kernels, layer 3 -------------------------
// H=1, C=40.

__global__ void edge_score_nc40_k(const float* __restrict__ xl,
                                  const float* __restrict__ xr,
                                  const int* __restrict__ eidx, int E, int E2,
                                  const float* __restrict__ att,   // [40]
                                  float* __restrict__ score,       // [E2]
                                  unsigned* __restrict__ mOrd) {   // [N]
  int e = blockIdx.x * blockDim.x + threadIdx.x;
  if (e >= E2) return;
  int src, dst;
  if (e < E) { src = eidx[e]; dst = eidx[E + e]; }
  else       { src = e - E;   dst = src; }
  const float4* a = (const float4*)(xl + (size_t)src * 40);
  const float4* b = (const float4*)(xr + (size_t)dst * 40);
  const float4* w = (const float4*)att;
  float p = 0.f;
#pragma unroll
  for (int q = 0; q < 10; ++q) {
    float4 av = a[q], bv = b[q], wv = w[q];
    float t;
    t = av.x + bv.x; t = t > 0.f ? t : NEG_SLOPE * t; p += wv.x * t;
    t = av.y + bv.y; t = t > 0.f ? t : NEG_SLOPE * t; p += wv.y * t;
    t = av.z + bv.z; t = t > 0.f ? t : NEG_SLOPE * t; p += wv.z * t;
    t = av.w + bv.w; t = t > 0.f ? t : NEG_SLOPE * t; p += wv.w * t;
  }
  score[e] = p;
  atomicMax(&mOrd[dst], f2ord(p));
}

__global__ void edge_agg_nc40_k(const float* __restrict__ xl,
                                const int* __restrict__ eidx, int E, int E2,
                                const float* __restrict__ score,
                                const float* __restrict__ ssum,
                                float* __restrict__ acc) {
  int tid = blockIdx.x * blockDim.x + threadIdx.x;
  if (tid >= E2 * 10) return;
  int e = tid / 10, q = tid - e * 10;
  int src, dst;
  if (e < E) { src = eidx[e]; dst = eidx[E + e]; }
  else       { src = e - E;   dst = src; }
  float alpha = score[e] / ssum[dst];
  float4 v = *(const float4*)(xl + (size_t)src * 40 + q * 4);
  float* o = acc + (size_t)dst * 40 + q * 4;
  unsafeAtomicAdd(o + 0, alpha * v.x);
  unsafeAtomicAdd(o + 1, alpha * v.y);
  unsafeAtomicAdd(o + 2, alpha * v.z);
  unsafeAtomicAdd(o + 3, alpha * v.w);
}

__global__ void final_bias_k(const float* __restrict__ acc,
                             const float* __restrict__ bias,
                             float* __restrict__ out, int n) {
  int t = blockIdx.x * blockDim.x + threadIdx.x;
  if (t < n) out[t] = acc[t] + bias[t % 40];
}

// ------------------------------- launcher ----------------------------------

extern "C" void kernel_launch(void* const* d_in, const int* in_sizes, int n_in,
                              void* d_out, int out_size, void* d_ws, size_t ws_size,
                              hipStream_t stream) {
  (void)n_in; (void)out_size; (void)ws_size;
  const float* x     = (const float*)d_in[0];
  const int*   eidx  = (const int*)  d_in[1];
  const float* W1l   = (const float*)d_in[2];  const float* b1l   = (const float*)d_in[3];
  const float* W1r   = (const float*)d_in[4];  const float* b1r   = (const float*)d_in[5];
  const float* att1  = (const float*)d_in[6];  const float* bias1 = (const float*)d_in[7];
  const float* W2l   = (const float*)d_in[8];  const float* b2l   = (const float*)d_in[9];
  const float* W2r   = (const float*)d_in[10]; const float* b2r   = (const float*)d_in[11];
  const float* att2  = (const float*)d_in[12]; const float* bias2 = (const float*)d_in[13];
  const float* W3l   = (const float*)d_in[14]; const float* b3l   = (const float*)d_in[15];
  const float* W3r   = (const float*)d_in[16]; const float* b3r   = (const float*)d_in[17];
  const float* att3  = (const float*)d_in[18]; const float* bias3 = (const float*)d_in[19];
  float* out = (float*)d_out;

  const int N  = in_sizes[0] / 128;   // 100000
  const int E  = in_sizes[1] / 2;     // 1000000
  const int E2 = E + N;               // + self loops

  // ---- carve workspace (256B aligned blocks) ----
  char* wp = (char*)d_ws;
  auto carve = [&](size_t bytes) -> void* {
    void* r = wp; wp += (bytes + 255) & ~(size_t)255; return r;
  };
  unsigned short* xbf = (unsigned short*)carve((size_t)N * 128 * 2);
  unsigned short* wA  = (unsigned short*)carve((size_t)128 * 128 * 2);
  unsigned short* wB  = (unsigned short*)carve((size_t)128 * 128 * 2);
  float*    xlb  = (float*)   carve((size_t)N * 128 * 4);
  float*    xrb  = (float*)   carve((size_t)N * 128 * 4);
  float*    scor = (float*)   carve((size_t)E2 * 4 * 4);
  float*    ssum = (float*)   carve((size_t)N * 4 * 4);
  unsigned* mOrd = (unsigned*)carve((size_t)N * 4 * 4);
  float*    acc  = (float*)   carve((size_t)N * 128 * 4);

  auto cdiv = [](long a, long b) { return (unsigned)((a + b - 1) / b); };
  const int rowTiles = (N + 15) / 16;

  // ---- layers 1 & 2 (HC=128 -> 8 col tiles = 2 groups of 4) ----
  auto run12 = [&](const float* hin, const float* Wl, const float* bl,
                   const float* Wr, const float* br,
                   const float* att, const float* bias) {
    cast_f32_bf16_k<<<cdiv((long)N * 128, 256), 256, 0, stream>>>(hin, xbf, N * 128);
    cast_f32_bf16_k<<<cdiv(128 * 128, 256), 256, 0, stream>>>(Wl, wA, 128 * 128);
    cast_f32_bf16_k<<<cdiv(128 * 128, 256), 256, 0, stream>>>(Wr, wB, 128 * 128);
    dim3 gg(cdiv(rowTiles, 4), 2);
    gemm_dual_bf16_wmma_k<4><<<gg, 128, 0, stream>>>(xbf, wA, wB, bl, br,
                                                     xlb, xrb, N, 128, rowTiles);
    fill_u32_k<<<cdiv((long)N * 4, 256), 256, 0, stream>>>(mOrd, ORD_NEG_INF, N * 4);
    fill_f32_k<<<cdiv((long)N * 4, 256), 256, 0, stream>>>(ssum, 0.f, N * 4);
    fill_f32_k<<<cdiv((long)N * 128, 256), 256, 0, stream>>>(acc, 0.f, N * 128);
    edge_score_h4_k<<<cdiv(E2, 4), 128, 0, stream>>>(xlb, xrb, eidx, E, E2, att, scor, mOrd);
    edge_exp_k<<<cdiv((long)E2 * 4, 256), 256, 0, stream>>>(scor, eidx, E, E2, 4, mOrd, ssum);
    edge_agg_h4_k<<<cdiv(E2, 4), 128, 0, stream>>>(xlb, eidx, E, E2, scor, ssum, acc);
    bias_elu_k<<<cdiv((long)N * 128, 256), 256, 0, stream>>>(acc, bias, N * 128);
  };

  run12(x,   W1l, b1l, W1r, b1r, att1, bias1);
  run12(acc, W2l, b2l, W2r, b2r, att2, bias2);

  // ---- layer 3 (H=1, C=40; weights zero-padded to 48 cols = 3 tiles) ----
  cast_f32_bf16_k<<<cdiv((long)N * 128, 256), 256, 0, stream>>>(acc, xbf, N * 128);
  fill_u32_k<<<cdiv(48 * 128 / 2, 256), 256, 0, stream>>>((unsigned*)wA, 0u, 48 * 128 / 2);
  fill_u32_k<<<cdiv(48 * 128 / 2, 256), 256, 0, stream>>>((unsigned*)wB, 0u, 48 * 128 / 2);
  cast_f32_bf16_k<<<cdiv(40 * 128, 256), 256, 0, stream>>>(W3l, wA, 40 * 128);
  cast_f32_bf16_k<<<cdiv(40 * 128, 256), 256, 0, stream>>>(W3r, wB, 40 * 128);
  dim3 g3(cdiv(rowTiles, 4), 1);
  gemm_dual_bf16_wmma_k<3><<<g3, 128, 0, stream>>>(xbf, wA, wB, b3l, b3r,
                                                   xlb, xrb, N, 40, rowTiles);
  fill_u32_k<<<cdiv(N, 256), 256, 0, stream>>>(mOrd, ORD_NEG_INF, N);
  fill_f32_k<<<cdiv(N, 256), 256, 0, stream>>>(ssum, 0.f, N);
  fill_f32_k<<<cdiv((long)N * 40, 256), 256, 0, stream>>>(acc, 0.f, N * 40);
  edge_score_nc40_k<<<cdiv(E2, 256), 256, 0, stream>>>(xlb, xrb, eidx, E, E2, att3, scor, mOrd);
  edge_exp_k<<<cdiv(E2, 256), 256, 0, stream>>>(scor, eidx, E, E2, 1, mOrd, ssum);
  edge_agg_nc40_k<<<cdiv((long)E2 * 10, 256), 256, 0, stream>>>(xlb, eidx, E, E2, scor, ssum, acc);
  final_bias_k<<<cdiv((long)N * 40, 256), 256, 0, stream>>>(acc, bias3, out, N * 40);
}
